// NeuroVoltron_32031866094389
// MI455X (gfx1250) — compile-verified, matrix operand order checked by reading, not run on hardware
//
#include <hip/hip_runtime.h>

// ---------------------------------------------------------------------------
// NeuroVoltron on gfx1250 (MI455X).
// Pass 1: pack loop-invariant weights into f16 WMMA B-fragments in d_ws.
// Pass 2: persistent per-batch-tile kernel runs the whole T-step recurrence.
//   - all GEMMs are v_wmma_f32_16x16x32_f16 tiles (batch rows = WMMA M dim)
//   - B-fragments: one aligned v16h global read (2x global_load_b128)
//   - A-fragments for z/h/hmid: f16 A-layout mirrors in LDS, written once in
//     the producing epilogue, read as one v16h (2x ds_load_b128), zero cvt.
// ---------------------------------------------------------------------------

typedef __attribute__((ext_vector_type(16))) _Float16 v16h;
typedef __attribute__((ext_vector_type(8)))  float    v8f;

#define BT   16   // batch tile (WMMA M)
#define NW   4    // waves per block
#define NTHR (NW * 32)
#define R_   12
#define L_   32
#define M_   16
#define HH_  32
#define HF_  64
#define E_   132
#define DT_  0.1f

// Fragment-id layout in workspace (each fragment = 512 halves = 1KB)
#define FID_MEAN(e)        (e)                         // 132
#define FID_ADD(e, nt)     (132 + (e) * 2 + (nt))      // 264
#define FID_GSW(r, nt)     (396 + (r) * 2 + (nt))      // 24
#define FID_GCW(r, nt)     (420 + (r) * 2 + (nt))      // 24
#define FID_LW(r, nt, kt)  (444 + (r) * 8 + (nt) * 2 + (kt)) // 96
#define FID_OW(r, nt, kt)  (540 + (r) * 4 + (nt) * 2 + (kt)) // 48
#define FID_WIH(r, nt)     (588 + (r) * 6 + (nt))      // 72
#define FID_WHH(r, nt)     (660 + (r) * 6 + (nt))      // 72
#define N_FRAGS            732
#define FRAG_HALVES        512

// Per-lane K index for 16-bit A/B fragments (ISA 7.12.2, 16-bit 16x32):
// lanes 0-15 hold K {0..7,16..23}; lanes 16-31 hold K {8..15,24..31}.
__device__ __forceinline__ int kmap(int s, int half) {
  return s + ((s & 8) ? 8 : 0) + half * 8;
}
// Inverse: element k -> (half, slot)
__device__ __forceinline__ int amap_half(int k) { return (k >> 3) & 1; }
__device__ __forceinline__ int amap_slot(int k) { return (k & 7) + ((k >> 4) << 3); }

// ---------------- pass 1: weight packing ----------------
__global__ __launch_bounds__(256)
void pack_weights_kernel(const float* __restrict__ mean_w, const float* __restrict__ add_w,
                         const float* __restrict__ gsw, const float* __restrict__ gcw,
                         const float* __restrict__ lw,  const float* __restrict__ ow,
                         const float* __restrict__ wih, const float* __restrict__ whh,
                         _Float16* __restrict__ wsf) {
  const int total = N_FRAGS * FRAG_HALVES;
  for (int idx = blockIdx.x * blockDim.x + threadIdx.x; idx < total;
       idx += gridDim.x * blockDim.x) {
    int fid  = idx >> 9;
    int r9   = idx & 511;
    int lane = r9 >> 4;
    int s    = r9 & 15;
    int half = lane >> 4;
    int n    = lane & 15;
    int k    = kmap(s, half);
    float v;
    if (fid < 132) {                      // mean_w (E,16,32), N=16 K=32
      int e = fid;
      v = mean_w[e * 512 + n * 32 + k];
    } else if (fid < 396) {               // add_w (E,32,16), N-tiles of 16, K=16
      int q = fid - 132, e = q >> 1, nt = q & 1;
      v = (k < 16) ? add_w[e * 512 + (nt * 16 + n) * 16 + k] : 0.0f;
    } else if (fid < 420) {               // gsw (R,32,32)
      int q = fid - 396, r = q >> 1, nt = q & 1;
      v = gsw[r * 1024 + (nt * 16 + n) * 32 + k];
    } else if (fid < 444) {               // gcw (R,32,32)
      int q = fid - 420, r = q >> 1, nt = q & 1;
      v = gcw[r * 1024 + (nt * 16 + n) * 32 + k];
    } else if (fid < 540) {               // lw (R,64,64): 4 n-tiles x 2 k-tiles
      int q = fid - 444, r = q >> 3, rem = q & 7, nt = rem >> 1, kt = rem & 1;
      v = lw[r * 4096 + (nt * 16 + n) * 64 + kt * 32 + k];
    } else if (fid < 588) {               // ow (R,32,64): 2 n-tiles x 2 k-tiles
      int q = fid - 540, r = q >> 2, rem = q & 3, nt = rem >> 1, kt = rem & 1;
      v = ow[r * 2048 + (nt * 16 + n) * 64 + kt * 32 + k];
    } else if (fid < 660) {               // wih (R,96,32): 6 n-tiles
      int q = fid - 588, r = q / 6, nt = q % 6;
      v = wih[r * 3072 + (nt * 16 + n) * 32 + k];
    } else {                              // whh (R,96,32): 6 n-tiles
      int q = fid - 660, r = q / 6, nt = q % 6;
      v = whh[r * 3072 + (nt * 16 + n) * 32 + k];
    }
    wsf[idx] = (_Float16)v;
  }
}

// ---------------- main kernel helpers ----------------
// A fragment built from f32 LDS: rows m = lane%16, base[m*ld + k] (k<kmax else 0)
__device__ __forceinline__ v16h frag_a(const float* base, int ld, int kmax, int lane) {
  int row = lane & 15, half = lane >> 4;
  v16h a;
#pragma unroll
  for (int s = 0; s < 16; ++s) {
    int k = kmap(s, half);
    a[s] = (k < kmax) ? (_Float16)base[row * ld + k] : (_Float16)0.0f;
  }
  return a;
}

// Pre-packed fragment: one aligned v16h per lane (2x b128 loads, no cvt)
__device__ __forceinline__ v16h frag_ld(const _Float16* base, int lane) {
  return ((const v16h*)base)[lane];
}

__device__ __forceinline__ v8f wmma_f16(v16h a, v16h b, v8f c) {
  return __builtin_amdgcn_wmma_f32_16x16x32_f16(false, a, false, b, (short)0, c,
                                                false, false);
}

__device__ __forceinline__ float sig_(float x) { return 1.0f / (1.0f + __expf(-x)); }

// In-wave LDS store->load ordering (same wave writes then re-reads scratch)
__device__ __forceinline__ void wave_lds_fence() {
  __builtin_amdgcn_wave_barrier();
  asm volatile("s_wait_dscnt 0" ::: "memory");
}

__global__ __launch_bounds__(NTHR)
void NeuroVoltron_kernel(const float* __restrict__ z0,  const float* __restrict__ h0,
                         const float* __restrict__ mean_b,
                         const float* __restrict__ gsb, const float* __restrict__ lb,
                         const float* __restrict__ ob,
                         const float* __restrict__ bih, const float* __restrict__ bhh,
                         const int* __restrict__ src_idx, const int* __restrict__ tgt_idx,
                         const int* __restrict__ nsteps,
                         const _Float16* __restrict__ wsf,
                         float* __restrict__ out) {
  // Persistent per-tile f32 state (R*L = 384 floats per batch row)
  __shared__ float sz[BT * 384];          // z
  __shared__ float sh[BT * 384];          // h
  __shared__ float sinc[BT * 384];        // inc_add (scatter target)
  __shared__ float swk[NW * 3072];        // per-wave scratch (overlaid phases)
  // f16 A-fragment mirrors: one 512-half fragment per node
  __shared__ _Float16 sz16[R_ * 512];     // z  (A layout)
  __shared__ _Float16 sh16[R_ * 512];     // h  (A layout)
  __shared__ _Float16 smean16[NW * 512];  // per-wave mean (A layout, K=16 pad 0)
  __shared__ int ssrc[E_], stgt[E_];

  const int tid  = threadIdx.x;
  const int wave = tid >> 5;
  const int lane = tid & 31;
  const int b0   = blockIdx.x * BT;
  const int T    = nsteps[0];

  const size_t zlen = (size_t)64 * T * R_ * L_;
  const size_t hoff = zlen;
  const size_t moff = zlen + (size_t)64 * R_ * HH_;

  // ---- init f32 state ----
  for (int i = tid; i < BT * 384; i += NTHR) {
    int m = i / 384, idx = i % 384;
    sz[i] = z0[(size_t)(b0 + m) * 384 + idx];
    sh[i] = h0[(size_t)(b0 + m) * 384 + idx];
  }
  // ---- init f16 A-layout mirrors (fragment element (lane',s) <- (m,k)) ----
  for (int i = tid; i < R_ * 512; i += NTHR) {
    int r = i >> 9, q = i & 511;
    int lp = q >> 4, s = q & 15;
    int m = lp & 15, half = lp >> 4;
    int k = kmap(s, half);
    sz16[i] = (_Float16)z0[(size_t)(b0 + m) * 384 + r * 32 + k];
    sh16[i] = (_Float16)h0[(size_t)(b0 + m) * 384 + r * 32 + k];
  }
  for (int i = tid; i < NW * 512; i += NTHR) smean16[i] = (_Float16)0.0f;
  for (int i = tid; i < E_; i += NTHR) { ssrc[i] = src_idx[i]; stgt[i] = tgt_idx[i]; }
  __syncthreads();

  // per-wave scratch overlays (floats)
  float*    wk      = &swk[wave * 3072];
  _Float16* whmid16 = (_Float16*)wk;  // 2 fragments (kt=0,1) = 1024 halves
  float*    wgate   = wk + 1024;      // 16x32 f32
  float*    wgi     = wk;             // 16x96 (GRU, overlays whmid16 after use)
  float*    wgh     = wk + 1536;      // 16x96
  _Float16* wmean16 = &smean16[wave * 512];

  const int n  = lane & 15;
  const int mb = (lane >> 4) * 8;

  for (int t = 0; t < T; ++t) {
    // ---- clear inc_add ----
    for (int i = tid; i < BT * 384; i += NTHR) sinc[i] = 0.0f;
    __syncthreads();

    // ================= edge message phase =================
    for (int e = wave; e < E_; e += NW) {
      const int s  = ssrc[e];
      const int tg = stgt[e];

      // prefetch next edge's fragment tiles (global_prefetch)
      if (e + NW < E_) {
        __builtin_prefetch(wsf + (size_t)FID_MEAN(e + NW) * FRAG_HALVES, 0, 0);
        __builtin_prefetch(wsf + (size_t)FID_ADD(e + NW, 0) * FRAG_HALVES, 0, 0);
      }

      // mean[16B x 16M] = z_src[16 x 32] * mean_w[e]^T
      v16h az = frag_ld(&sz16[s * 512], lane);
      v16h bw = frag_ld(wsf + (size_t)FID_MEAN(e) * FRAG_HALVES, lane);
      v8f  c  = {};
      c = wmma_f16(az, bw, c);

      const float bias = mean_b[e * M_ + n];
      const int   hn   = amap_half(n), sn = amap_slot(n);  // k=n (K=16 frag)
#pragma unroll
      for (int j = 0; j < 8; ++j) {
        int   m = mb + j;
        float v = c[j] + bias;
        wmean16[(m + 16 * hn) * 16 + sn] = (_Float16)v;
        out[moff + ((size_t)(b0 + m) * T + t) * (E_ * M_) + e * M_ + n] = v;
      }
      wave_lds_fence();

      // add[16B x 32L] = mean[16 x 16(pad32)] * add_w[e]^T, scatter into inc[tgt]
      v16h am = frag_ld(wmean16, lane);
#pragma unroll
      for (int nt = 0; nt < 2; ++nt) {
        v16h bb = frag_ld(wsf + (size_t)FID_ADD(e, nt) * FRAG_HALVES, lane);
        v8f  cc = {};
        cc = wmma_f16(am, bb, cc);
#pragma unroll
        for (int j = 0; j < 8; ++j) {
          int m = mb + j;
          atomicAdd(&sinc[m * 384 + tg * L_ + nt * 16 + n], cc[j]);
        }
      }
    }
    __syncthreads();

    // ================= per-node flow + GRU phase =================
    for (int r = wave; r < R_; r += NW) {
      v16h az = frag_ld(&sz16[r * 512], lane);
      v16h ai = frag_a(&sinc[r * L_], 384, L_, lane);

      // gate = sigmoid(z*gsw^T + inc*gcw^T + gsb)
#pragma unroll
      for (int nt = 0; nt < 2; ++nt) {
        v16h b1 = frag_ld(wsf + (size_t)FID_GSW(r, nt) * FRAG_HALVES, lane);
        v16h b2 = frag_ld(wsf + (size_t)FID_GCW(r, nt) * FRAG_HALVES, lane);
        v8f  c  = {};
        c = wmma_f16(az, b1, c);
        c = wmma_f16(ai, b2, c);
#pragma unroll
        for (int j = 0; j < 8; ++j) {
          int o = nt * 16 + n;
          wgate[(mb + j) * L_ + o] = sig_(c[j] + gsb[r * L_ + o]);
        }
      }

      // hmid = silu([z,inc] * lw^T + lb)  -> stored directly in f16 A layout
#pragma unroll
      for (int nt = 0; nt < 4; ++nt) {
        v16h b1 = frag_ld(wsf + (size_t)FID_LW(r, nt, 0) * FRAG_HALVES, lane);
        v16h b2 = frag_ld(wsf + (size_t)FID_LW(r, nt, 1) * FRAG_HALVES, lane);
        v8f  c  = {};
        c = wmma_f16(az, b1, c);
        c = wmma_f16(ai, b2, c);
        const int f0 = nt * 16 + n;           // feature index 0..63
        const int kt = nt >> 1;               // k-tile of the A fragment
        const int kk = f0 & 31;               // k within fragment
        const int hh = amap_half(kk), ss = amap_slot(kk);
#pragma unroll
        for (int j = 0; j < 8; ++j) {
          int   m = mb + j;
          float v = c[j] + lb[r * HF_ + f0];
          whmid16[kt * 512 + (m + 16 * hh) * 16 + ss] = (_Float16)(v * sig_(v));
        }
      }
      wave_lds_fence();

      // target = tanh(hmid * ow^T + ob); z_new = z + DT*gate*(target - z)
      v16h ah0 = frag_ld(whmid16, lane);
      v16h ah1 = frag_ld(whmid16 + 512, lane);
#pragma unroll
      for (int nt = 0; nt < 2; ++nt) {
        v16h b1 = frag_ld(wsf + (size_t)FID_OW(r, nt, 0) * FRAG_HALVES, lane);
        v16h b2 = frag_ld(wsf + (size_t)FID_OW(r, nt, 1) * FRAG_HALVES, lane);
        v8f  c  = {};
        c = wmma_f16(ah0, b1, c);
        c = wmma_f16(ah1, b2, c);
        const int l  = nt * 16 + n;
        const int hl = amap_half(l), sl = amap_slot(l);
#pragma unroll
        for (int j = 0; j < 8; ++j) {
          int   m  = mb + j;
          float tv = tanhf(c[j] + ob[r * L_ + l]);
          float zv = sz[m * 384 + r * L_ + l];
          float g  = wgate[m * L_ + l];
          float zn = zv + DT_ * g * (tv - zv);
          sz[m * 384 + r * L_ + l]              = zn;
          sz16[r * 512 + (m + 16 * hl) * 16 + sl] = (_Float16)zn;
          out[((size_t)(b0 + m) * T + t) * (R_ * L_) + r * L_ + l] = zn;
        }
      }
      wave_lds_fence();  // wgate/whmid16 dead; wgi/wgh overlay them below

      // GRU: gi = inc*wih^T + bih ; gh = h*whh^T + bhh
      v16h ah = frag_ld(&sh16[r * 512], lane);
#pragma unroll
      for (int nt = 0; nt < 6; ++nt) {
        v16h b1 = frag_ld(wsf + (size_t)FID_WIH(r, nt) * FRAG_HALVES, lane);
        v8f  c1 = {};
        c1 = wmma_f16(ai, b1, c1);
        v16h b2 = frag_ld(wsf + (size_t)FID_WHH(r, nt) * FRAG_HALVES, lane);
        v8f  c2 = {};
        c2 = wmma_f16(ah, b2, c2);
#pragma unroll
        for (int j = 0; j < 8; ++j) {
          int o = nt * 16 + n;
          wgi[(mb + j) * 96 + o] = c1[j] + bih[r * 96 + o];
          wgh[(mb + j) * 96 + o] = c2[j] + bhh[r * 96 + o];
        }
      }
      wave_lds_fence();

      // GRU elementwise combine (16 batch rows x 32 hidden), update f32+f16 h
      for (int idx = lane; idx < BT * HH_; idx += 32) {
        int   m  = idx >> 5, cc = idx & 31;
        float ir = wgi[m * 96 + cc], iz = wgi[m * 96 + 32 + cc], in_ = wgi[m * 96 + 64 + cc];
        float hr = wgh[m * 96 + cc], hz = wgh[m * 96 + 32 + cc], hn  = wgh[m * 96 + 64 + cc];
        float hv = sh[m * 384 + r * L_ + cc];
        float rs = sig_(ir + hr);
        float up = sig_(iz + hz);
        float nn = tanhf(in_ + rs * hn);
        float hnew = (1.0f - up) * nn + up * hv;
        sh[m * 384 + r * L_ + cc] = hnew;
        sh16[r * 512 + (m + 16 * amap_half(cc)) * 16 + amap_slot(cc)] = (_Float16)hnew;
      }
      wave_lds_fence();
    }
    __syncthreads();
  }

  // ---- final h ----
  for (int i = tid; i < BT * 384; i += NTHR) {
    int m = i / 384, idx = i % 384;
    out[hoff + (size_t)(b0 + m) * 384 + idx] = sh[i];
  }
}

extern "C" void kernel_launch(void* const* d_in, const int* in_sizes, int n_in,
                              void* d_out, int out_size, void* d_ws, size_t ws_size,
                              hipStream_t stream) {
  (void)in_sizes; (void)n_in; (void)out_size; (void)ws_size;
  _Float16* wsf = (_Float16*)d_ws;

  // Pass 1: pack weights into WMMA B-fragment layout (f16) in workspace.
  {
    const int total  = N_FRAGS * FRAG_HALVES;
    const int blocks = (total + 255) / 256;
    pack_weights_kernel<<<dim3(blocks), dim3(256), 0, stream>>>(
        (const float*)d_in[2],  /* mean_w */
        (const float*)d_in[4],  /* add_w  */
        (const float*)d_in[5],  /* gsw    */
        (const float*)d_in[7],  /* gcw    */
        (const float*)d_in[8],  /* lw     */
        (const float*)d_in[10], /* ow     */
        (const float*)d_in[12], /* wih    */
        (const float*)d_in[13], /* whh    */
        wsf);
  }

  // Pass 2: full recurrence.
  NeuroVoltron_kernel<<<dim3(64 / BT), dim3(NTHR), 0, stream>>>(
      (const float*)d_in[0],  /* z0     */
      (const float*)d_in[1],  /* h0     */
      (const float*)d_in[3],  /* mean_b */
      (const float*)d_in[6],  /* gsb    */
      (const float*)d_in[9],  /* lb     */
      (const float*)d_in[11], /* ob     */
      (const float*)d_in[14], /* bih    */
      (const float*)d_in[15], /* bhh    */
      (const int*)d_in[16],   /* src    */
      (const int*)d_in[17],   /* tgt    */
      (const int*)d_in[18],   /* n_steps*/
      wsf, (float*)d_out);
}
